// Grid_Attention_35132832481760
// MI455X (gfx1250) — matrix operand order repaired and checked by reading, not compiled, not run on hardware
//
#include <hip/hip_runtime.h>

typedef __attribute__((ext_vector_type(16))) _Float16 v16h;
typedef __attribute__((ext_vector_type(8)))  _Float16 v8h;
typedef __attribute__((ext_vector_type(8)))  float    v8f;

#define WIN_N   49
#define DIMK    256
#define N_HEADS 8
#define DHEAD   32
#define XS_STRIDE 272   // halfs per xs row (256 + 16 pad, keeps 16B alignment)
#define PS_STRIDE 64    // halfs per P row
#define ATTN_SCALE 0.17677669529663687f  // 32^-0.5

// A-fragment (16x32 f16, MxK): lane m = (lane&15) within tile; elements:
//   e=0..7  -> K = k0 + hi*8 + e ;  e=8..15 -> K = k0 + 16 + hi*8 + (e-8)
__device__ __forceinline__ v16h frag_a(const _Float16* rowbase, int k0, int hi) {
  const v8h lo = *(const v8h*)(rowbase + k0 + hi * 8);
  const v8h hh = *(const v8h*)(rowbase + k0 + hi * 8 + 16);
  return __builtin_shufflevector(lo, hh, 0,1,2,3,4,5,6,7, 8,9,10,11,12,13,14,15);
}

// B-fragment (32x16 f16, KxN): lane n = (lane&15) within tile; elements:
//   e=0..7  -> K = k0 + hi*8 + e ;  e=8..15 -> K = k0 + 16 + hi*8 + (e-8)
__device__ __forceinline__ v16h frag_b(const _Float16* rowbase, int k0, int hi) {
  const v8h lo = *(const v8h*)(rowbase + k0 + hi * 8);
  const v8h hh = *(const v8h*)(rowbase + k0 + hi * 8 + 16);
  return __builtin_shufflevector(lo, hh, 0,1,2,3,4,5,6,7, 8,9,10,11,12,13,14,15);
}

// One-shot pre-pass: w_qkv f32 [768*256] -> f16 copy in workspace (L2-resident).
__global__ void __launch_bounds__(256)
convert_w_kernel(const float* __restrict__ w, _Float16* __restrict__ wh) {
  const int t = blockIdx.x * 256 + threadIdx.x;       // one float4 per thread
  if (t < (3 * DIMK * DIMK) / 4) {
    const float4 f = ((const float4*)w)[t];
    _Float16* d = wh + t * 4;
    d[0] = (_Float16)f.x; d[1] = (_Float16)f.y;
    d[2] = (_Float16)f.z; d[3] = (_Float16)f.w;
  }
}

__global__ void __launch_bounds__(256)
win_attn_kernel(const float* __restrict__ x, const _Float16* __restrict__ wh,
                const float* __restrict__ bias_table, const int* __restrict__ rel_index,
                float* __restrict__ out) {
  extern __shared__ char smem[];
  const int tid  = threadIdx.x;
  const int wave = tid >> 5;     // == head index
  const int lane = tid & 31;
  const int l15  = lane & 15;
  const int hi   = lane >> 4;
  const int h    = wave;
  const size_t win_base = (size_t)blockIdx.x * (WIN_N * DIMK);

  // LDS layout (160 KB dynamic):
  //   [0, 64K)      : xs[64][272] f16 window, rows 49..63 zeroed
  //                   (aliased later by per-wave Ps[64][64])
  //   [64K, 96K)    : per-wave Qs[64][32] f16
  //   [96K, 128K)   : per-wave Ks[64][32] f16
  //   [128K, 160K)  : per-wave Vt[32][64] f16 (V transposed)
  _Float16* xs = (_Float16*)smem;
  _Float16* Ps = (_Float16*)smem + wave * (64 * PS_STRIDE);
  _Float16* Qs = (_Float16*)(smem + 65536)  + wave * (64 * 32);
  _Float16* Ks = (_Float16*)(smem + 98304)  + wave * (64 * 32);
  _Float16* Vt = (_Float16*)(smem + 131072) + wave * (32 * 64);

  // ---- Zero-pad rows 49..63 (so A-fragment loads need no masking) ----
  for (int t = tid; t < 15 * (DIMK / 4); t += 256) {
    const int row = 49 + t / (DIMK / 4);
    const int d   = (t % (DIMK / 4)) * 4;
    _Float16* dst = xs + row * XS_STRIDE + d;
    dst[0] = (_Float16)0.f; dst[1] = (_Float16)0.f;
    dst[2] = (_Float16)0.f; dst[3] = (_Float16)0.f;
  }

  // ---- Stage contiguous window [49][256] f32 -> LDS f16 ----
  const float4* xv = (const float4*)(x + win_base);
  for (int t = tid; t < (WIN_N * DIMK) / 4; t += 256) {
    const float4 f = xv[t];
    const int e   = t * 4;
    const int tok = e >> 8;
    const int d   = e & 255;
    _Float16* dst = xs + tok * XS_STRIDE + d;
    dst[0] = (_Float16)f.x; dst[1] = (_Float16)f.y;
    dst[2] = (_Float16)f.z; dst[3] = (_Float16)f.w;
  }
  __syncthreads();

  // ---- QKV projection: [64x256]x[256x32] per head, per q/k/v (f16 weights) ----
  for (int qkv = 0; qkv < 3; ++qkv) {
    const int cbase = qkv * DIMK + h * DHEAD;
    // Destination selected once per qkv (no per-element branches):
    //   Qs/Ks: row-major [token][chan]  -> addr = row*32 + col
    //   Vt   : transposed [chan][token] -> addr = col*64 + row
    _Float16* dstbase = (qkv == 0) ? Qs : (qkv == 1) ? Ks : Vt;
    const int rstride = (qkv == 2) ? 1 : 32;
    const int cstride = (qkv == 2) ? 64 : 1;
    for (int mo = 0; mo < 4; ++mo) {
      const _Float16* xrow = xs + (mo * 16 + l15) * XS_STRIDE;
#pragma unroll
      for (int no = 0; no < 2; ++no) {
        const _Float16* wrow = wh + (size_t)(cbase + no * 16 + l15) * DIMK;
        v8f acc = {};
#pragma unroll
        for (int kk = 0; kk < 8; ++kk) {
          const v16h a = frag_a(xrow, kk * 32, hi);
          const v16h b = frag_b(wrow, kk * 32, hi);
          acc = __builtin_amdgcn_wmma_f32_16x16x32_f16(false, a, false, b,
                                                       (short)0, acc, false, false);
        }
        // C/D layout: element v -> token row = mo*16 + v + 8*hi ; col = no*16+l15
        _Float16* dst = dstbase + (mo * 16 + 8 * hi) * rstride
                                + (no * 16 + l15) * cstride;
#pragma unroll
        for (int v = 0; v < 8; ++v)
          dst[v * rstride] = (_Float16)acc[v];
      }
    }
  }
  __syncthreads();  // everyone done reading xs before Ps (aliased) is written

  // ---- S = Q K^T * scale + bias ; softmax ; P -> LDS (f16) ----
  for (int mo = 0; mo < 4; ++mo) {
    const v16h aQ = frag_a(Qs + (mo * 16 + l15) * 32, 0, hi);
    v8f s[4];
#pragma unroll
    for (int no = 0; no < 4; ++no) {
      const v16h bK = frag_b(Ks + (no * 16 + l15) * 32, 0, hi);
      v8f z = {};
      s[no] = __builtin_amdgcn_wmma_f32_16x16x32_f16(false, aQ, false, bK,
                                                     (short)0, z, false, false);
    }
#pragma unroll
    for (int v = 0; v < 8; ++v) {
      const int i = mo * 16 + v + 8 * hi;
      float vals[4];
      float rmax = -3.0e38f;
#pragma unroll
      for (int no = 0; no < 4; ++no) {
        const int j = no * 16 + l15;
        float sv = -1.0e30f;
        if (i < WIN_N && j < WIN_N)
          sv = s[no][v] * ATTN_SCALE +
               bias_table[rel_index[i * WIN_N + j] * N_HEADS + h];
        vals[no] = sv;
        rmax = fmaxf(rmax, sv);
      }
#pragma unroll
      for (int off = 8; off >= 1; off >>= 1)
        rmax = fmaxf(rmax, __shfl_xor(rmax, off, 32));
      float ev[4];
      float rsum = 0.f;
#pragma unroll
      for (int no = 0; no < 4; ++no) {
        const float e = (vals[no] > -1.0e29f) ? __expf(vals[no] - rmax) : 0.f;
        ev[no] = e; rsum += e;
      }
#pragma unroll
      for (int off = 8; off >= 1; off >>= 1)
        rsum += __shfl_xor(rsum, off, 32);
      const float inv = (rsum > 0.f) ? (1.f / rsum) : 0.f;
#pragma unroll
      for (int no = 0; no < 4; ++no)
        Ps[i * PS_STRIDE + no * 16 + l15] = (_Float16)(ev[no] * inv);
    }
  }

  // ---- O = P V ; write to contiguous output ----
  for (int mo = 0; mo < 4; ++mo) {
#pragma unroll
    for (int no = 0; no < 2; ++no) {
      v8f acc = {};
#pragma unroll
      for (int kk = 0; kk < 2; ++kk) {
        const v16h aP = frag_a(Ps + (mo * 16 + l15) * PS_STRIDE, kk * 32, hi);
        const v16h bV = frag_b(Vt + (no * 16 + l15) * 64, kk * 32, hi);
        acc = __builtin_amdgcn_wmma_f32_16x16x32_f16(false, aP, false, bV,
                                                     (short)0, acc, false, false);
      }
#pragma unroll
      for (int v = 0; v < 8; ++v) {
        const int i = mo * 16 + v + 8 * hi;
        if (i < WIN_N)
          out[win_base + (size_t)i * DIMK + h * DHEAD + no * 16 + l15] = acc[v];
      }
    }
  }
}

extern "C" void kernel_launch(void* const* d_in, const int* in_sizes, int n_in,
                              void* d_out, int out_size, void* d_ws, size_t ws_size,
                              hipStream_t stream) {
  const float* x  = (const float*)d_in[0];
  const float* w  = (const float*)d_in[1];
  const float* bt = (const float*)d_in[2];
  const int*   ri = (const int*)d_in[3];
  float* out = (float*)d_out;
  _Float16* wh = (_Float16*)d_ws;                  // 3*256*256*2 = 393216 B scratch

  // One-shot (deterministic, repeated each call) f32->f16 weight conversion.
  const int wq4 = (3 * DIMK * DIMK) / 4;
  convert_w_kernel<<<dim3((wq4 + 255) / 256), dim3(256), 0, stream>>>(w, wh);

  const int nwin = in_sizes[0] / (WIN_N * DIMK);   // b*X*Y windows
  const size_t smem = 65536 + 3 * 32768;           // 160 KB dynamic LDS
  win_attn_kernel<<<dim3(nwin), dim3(256), smem, stream>>>(x, wh, bt, ri, out);
}
